// PokeNet_4801773437053
// MI455X (gfx1250) — matrix operand order repaired and checked by reading, not compiled, not run on hardware
//
#include <hip/hip_runtime.h>

typedef _Float16 h16;
typedef __attribute__((ext_vector_type(8)))  _Float16 v8h;
typedef __attribute__((ext_vector_type(16))) _Float16 v16h;
typedef __attribute__((ext_vector_type(8)))  float    v8f;

#define LRELU_NEG 0.01f
__device__ __forceinline__ float lrelu(float x) { return x >= 0.f ? x : LRELU_NEG * x; }

// ------------------------------------------------------------------
// Scatter points into dense f32 volume + mask (last-write-wins, like .set)
// ------------------------------------------------------------------
__global__ void k_scatter(const float* __restrict__ feat, const int* __restrict__ coors,
                          float* __restrict__ dense, float* __restrict__ mask, int n) {
  int i = blockIdx.x * blockDim.x + threadIdx.x;
  if (i >= n) return;
  int b = coors[4*i+0], z = coors[4*i+1], y = coors[4*i+2], x = coors[4*i+3];
  long idx = (((long)b*50 + z)*60 + y)*65 + x;
  dense[idx] = feat[i];
  mask[idx]  = 1.0f;
}

// ------------------------------------------------------------------
// Weight reorder: w[co][ci][4][4][4] (f32) -> wr[tap][co][ci] (f16)
// ------------------------------------------------------------------
__global__ void k_reorder32(const float* __restrict__ w, h16* __restrict__ wr) {
  int t = blockIdx.x * blockDim.x + threadIdx.x;
  if (t >= 64*32*32) return;
  int ci = t & 31, co = (t >> 5) & 31, tap = t >> 10;
  wr[t] = (h16)w[(co*32 + ci)*64 + tap];
}
__global__ void k_reorder7(const float* __restrict__ w, h16* __restrict__ wr) {
  int t = blockIdx.x * blockDim.x + threadIdx.x;
  if (t >= 2048) return;
  int ci = t & 31, tap = t >> 5;
  wr[t] = (h16)w[ci*64 + tap];          // w7 shape (1,32,4,4,4)
}

// ------------------------------------------------------------------
// Generic 1D max-pool over one axis, f32 single-channel volume.
// pm: optional pre-mask at INPUT dims (val -> -inf where pm<=0)
// gm: optional gate mask at OUTPUT dims (out -> 0 where gm<=0)
// ------------------------------------------------------------------
__global__ void k_pool1d_f32(const float* __restrict__ in, float* __restrict__ out,
                             const float* __restrict__ pm, const float* __restrict__ gm,
                             int NB, int Di, int Hi, int Wi, int axis, int k) {
  int Do = Di - (axis == 0 ? k-1 : 0);
  int Ho = Hi - (axis == 1 ? k-1 : 0);
  int Wo = Wi - (axis == 2 ? k-1 : 0);
  long total = (long)NB * Do * Ho * Wo;
  long t = blockIdx.x * (long)blockDim.x + threadIdx.x;
  if (t >= total) return;
  int x = (int)(t % Wo); long r = t / Wo;
  int y = (int)(r % Ho); r /= Ho;
  int z = (int)(r % Do); int n = (int)(r / Do);
  long base = (((long)n*Di + z)*Hi + y)*Wi + x;
  long stride = (axis == 2) ? 1 : (axis == 1) ? (long)Wi : (long)Hi*Wi;
  float m = -INFINITY;
  for (int j = 0; j < k; j++) {
    long ii = base + (long)j * stride;
    float v = in[ii];
    if (pm && pm[ii] <= 0.f) v = -INFINITY;
    m = fmaxf(m, v);
  }
  if (gm) m = (gm[t] > 0.f) ? m : 0.f;
  out[t] = m;
}

// ------------------------------------------------------------------
// Generic 1D max-pool over one axis, channels-last f16 (C=32).
// One thread = one output voxel * 8-channel group.
// ------------------------------------------------------------------
__global__ void k_pool1d_f16c(const h16* __restrict__ in, h16* __restrict__ out,
                              const float* __restrict__ pm, const float* __restrict__ gm,
                              int NB, int Di, int Hi, int Wi, int axis, int k) {
  int Do = Di - (axis == 0 ? k-1 : 0);
  int Ho = Hi - (axis == 1 ? k-1 : 0);
  int Wo = Wi - (axis == 2 ? k-1 : 0);
  long npos = (long)NB * Do * Ho * Wo;
  long t = blockIdx.x * (long)blockDim.x + threadIdx.x;
  if (t >= npos * 4) return;
  int g = (int)(t & 3);
  long p = t >> 2;
  int x = (int)(p % Wo); long r = p / Wo;
  int y = (int)(r % Ho); r /= Ho;
  int z = (int)(r % Do); int n = (int)(r / Do);
  long base = (((long)n*Di + z)*Hi + y)*Wi + x;
  long stride = (axis == 2) ? 1 : (axis == 1) ? (long)Wi : (long)Hi*Wi;
  float acc[8];
#pragma unroll
  for (int c = 0; c < 8; c++) acc[c] = -INFINITY;
  for (int j = 0; j < k; j++) {
    long ip = base + (long)j * stride;
    if (pm && pm[ip] <= 0.f) continue;
    const h16* q = in + ip*32 + g*8;
#pragma unroll
    for (int c = 0; c < 8; c++) acc[c] = fmaxf(acc[c], (float)q[c]);
  }
  float gv = gm ? gm[p] : 1.f;
  h16* o = out + p*32 + g*8;
#pragma unroll
  for (int c = 0; c < 8; c++) {
    float v = acc[c];
    if (gm) v = (gv > 0.f) ? v : 0.f;
    o[c] = (h16)v;
  }
}

// ------------------------------------------------------------------
// conv1: 1 -> 32 channels, 4^3 VALID, scalar (only 2.7 GMAC).
// ------------------------------------------------------------------
__global__ void k_conv1(const float* __restrict__ in, const float* __restrict__ m2,
                        const float* __restrict__ w1, const float* __restrict__ b1,
                        h16* __restrict__ out) {
  const int Di = 50, Hi = 60, Wi = 65, Do = 47, Ho = 57, Wo = 62;
  long total = (long)8 * Do * Ho * Wo;
  long t = blockIdx.x * (long)blockDim.x + threadIdx.x;
  if (t >= total) return;
  int x = (int)(t % Wo); long r = t / Wo;
  int y = (int)(r % Ho); r /= Ho;
  int z = (int)(r % Do); int n = (int)(r / Do);
  h16* o = out + t*32;
  float mv = m2[t];
  if (mv <= 0.f) {
#pragma unroll
    for (int c = 0; c < 32; c++) o[c] = (h16)0.f;
    return;
  }
  float acc[32];
#pragma unroll
  for (int c = 0; c < 32; c++) acc[c] = 0.f;
  int tap = 0;
  for (int kz = 0; kz < 4; kz++)
    for (int ky = 0; ky < 4; ky++)
      for (int kx = 0; kx < 4; kx++, tap++) {
        float v = in[(((long)n*Di + z + kz)*Hi + y + ky)*Wi + x + kx];
#pragma unroll
        for (int c = 0; c < 32; c++) acc[c] += v * w1[c*64 + tap];
      }
#pragma unroll
  for (int c = 0; c < 32; c++) o[c] = (h16)lrelu(acc[c] + b1[c]);
}

// ------------------------------------------------------------------
// 32->32 conv, 4^3 VALID, implicit GEMM on v_wmma_f32_16x16x32_f16.
// Weights (128 KB) staged in LDS (WGP has 320 KB; working set > 64 KB WGP$).
// One wave = 2 y-rows x 16 x-positions x 32 output channels:
// 4 accumulators, A loaded per row, B loaded once from LDS, 4 WMMA/tap.
// ------------------------------------------------------------------
__global__ void __launch_bounds__(256)
k_conv_wmma(const h16* __restrict__ in, h16* __restrict__ out,
            const h16* __restrict__ wr, const float* __restrict__ bias,
            const float* __restrict__ m2, int Di, int Hi, int Wi) {
  const int Do = Di - 3, Ho = Hi - 3, Wo = Wi - 3;
  const int nxt = (Wo + 15) >> 4;
  const int nyt = (Ho + 1) >> 1;

  extern __shared__ h16 sw[];                 // 64*32*32 halves = 128 KB
  {
    const v8h* s = (const v8h*)wr;
    v8h* d = (v8h*)sw;
    for (int i = threadIdx.x; i < 8192; i += 256) d[i] = s[i];
  }
  __syncthreads();                            // all threads reach this before any exit

  long wave = (blockIdx.x * (long)blockDim.x + threadIdx.x) >> 5;
  long totalw = (long)8 * Do * nyt * nxt;
  if (wave >= totalw) return;                 // wave-uniform exit (EXEC stays full)
  int lane = threadIdx.x & 31;
  int half = lane >> 4, l = lane & 15;
  int xt = (int)(wave % nxt); long r = wave / nxt;
  int yt = (int)(r % nyt); r /= nyt;
  int zo = (int)(r % Do); int n = (int)(r / Do);
  int x0 = xt << 4;
  int y0 = yt << 1;

  v8f acc[2][2];
#pragma unroll
  for (int i = 0; i < 2; i++)
#pragma unroll
    for (int j = 0; j < 2; j++) acc[i][j] = (v8f){0.f,0.f,0.f,0.f,0.f,0.f,0.f,0.f};

  // per-lane A row base: spatial x = x0 + (lane&15), channels contiguous
  const h16* arow = in + ((((long)n*Di + zo)*Hi + y0)*Wi + (x0 + l)) * 32;
  const long yst = (long)Wi * 32;

  for (int kz = 0; kz < 4; kz++)
    for (int ky = 0; ky < 4; ky++) {
      const h16* r0 = arow + ((long)kz*Hi + ky) * yst;
      const h16* r1 = r0 + yst;
#pragma unroll
      for (int kx = 0; kx < 4; kx++) {
        const h16* p0 = r0 + kx*32;
        const h16* p1 = r1 + kx*32;
        // A k order used here (contiguous, same as B): lanes<16 hold K=0..15; lanes>=16 K=16..31
        v8h a0 = *(const v8h*)(p0 + half*16);
        v8h a1 = *(const v8h*)(p0 + half*16 + 8);
        v16h A0 = __builtin_shufflevector(a0, a1, 0,1,2,3,4,5,6,7,8,9,10,11,12,13,14,15);
        v8h c0 = *(const v8h*)(p1 + half*16);
        v8h c1 = *(const v8h*)(p1 + half*16 + 8);
        v16h A1 = __builtin_shufflevector(c0, c1, 0,1,2,3,4,5,6,7,8,9,10,11,12,13,14,15);
        int tap = (kz*4 + ky)*4 + kx;
        // B from LDS, same contiguous k order as A: lane n<16 -> col N=n, K=0..15; lane>=16 -> N=lane-16, K=16..31
        const h16* wb = sw + tap*1024 + half*16;
        v16h B0 = *(const v16h*)(wb + l*32);          // co in [0,16)
        v16h B1 = *(const v16h*)(wb + (16 + l)*32);   // co in [16,32)
        acc[0][0] = __builtin_amdgcn_wmma_f32_16x16x32_f16(false, A0, false, B0, (short)0, acc[0][0], false, false);
        acc[0][1] = __builtin_amdgcn_wmma_f32_16x16x32_f16(false, A0, false, B1, (short)0, acc[0][1], false, false);
        acc[1][0] = __builtin_amdgcn_wmma_f32_16x16x32_f16(false, A1, false, B0, (short)0, acc[1][0], false, false);
        acc[1][1] = __builtin_amdgcn_wmma_f32_16x16x32_f16(false, A1, false, B1, (short)0, acc[1][1], false, false);
      }
    }

  // Epilogue: C/D layout -> lane<16: (M=v, N=l); lane>=16: (M=v+8, N=l)
#pragma unroll
  for (int row = 0; row < 2; row++) {
    int y = y0 + row;
    if (y < Ho) {
      long obase = (((long)n*Do + zo)*Ho + y) * Wo;
#pragma unroll
      for (int v = 0; v < 8; v++) {
        int x = x0 + v + half*8;
        if (x < Wo) {
          long op = obase + x;
          float mv = m2[op];
          float o0 = (mv > 0.f) ? lrelu(acc[row][0][v] + bias[l])      : 0.f;
          float o1 = (mv > 0.f) ? lrelu(acc[row][1][v] + bias[16 + l]) : 0.f;
          out[op*32 + l]      = (h16)o0;
          out[op*32 + 16 + l] = (h16)o1;
        }
      }
    }
  }
}

// ------------------------------------------------------------------
// conv7: 32 -> 1 channel, scalar (0.4 GMAC). out f32, single channel.
// ------------------------------------------------------------------
__global__ void k_conv7(const h16* __restrict__ in, float* __restrict__ out,
                        const h16* __restrict__ w7r, const float* __restrict__ b7,
                        const float* __restrict__ m2, int Di, int Hi, int Wi) {
  const int Do = Di - 3, Ho = Hi - 3, Wo = Wi - 3;
  long total = (long)8 * Do * Ho * Wo;
  long t = blockIdx.x * (long)blockDim.x + threadIdx.x;
  if (t >= total) return;
  int x = (int)(t % Wo); long r = t / Wo;
  int y = (int)(r % Ho); r /= Ho;
  int z = (int)(r % Do); int n = (int)(r / Do);
  if (m2[t] <= 0.f) { out[t] = 0.f; return; }
  float acc = 0.f;
  int tap = 0;
  for (int kz = 0; kz < 4; kz++)
    for (int ky = 0; ky < 4; ky++)
      for (int kx = 0; kx < 4; kx++, tap++) {
        const h16* p = in + ((((long)n*Di + z + kz)*Hi + y + ky)*Wi + x + kx)*32;
        const h16* wq = w7r + tap*32;
#pragma unroll
        for (int ci = 0; ci < 32; ci++) acc += (float)p[ci] * (float)wq[ci];
      }
  out[t] = lrelu(acc + b7[0]);
}

// ------------------------------------------------------------------
// FC: out[n][j] = lrelu(sum_i h[n][i] * W[i][j] + b[j]); thread per j,
// consecutive j -> coalesced W reads; h broadcast from L2.
// ------------------------------------------------------------------
__global__ void k_fc(const float* __restrict__ h, const float* __restrict__ W,
                     const float* __restrict__ b, float* __restrict__ out,
                     int K, int Nout) {
  int j = blockIdx.x * blockDim.x + threadIdx.x;
  if (j >= Nout) return;
  float acc[8];
#pragma unroll
  for (int n = 0; n < 8; n++) acc[n] = 0.f;
  for (int i = 0; i < K; i++) {
    float w = W[(long)i*Nout + j];
#pragma unroll
    for (int n = 0; n < 8; n++) acc[n] += h[(long)n*K + i] * w;
  }
#pragma unroll
  for (int n = 0; n < 8; n++) out[n*Nout + j] = lrelu(acc[n] + b[j]);
}

__global__ void k_fc_last(const float* __restrict__ h, const float* __restrict__ W,
                          const float* __restrict__ b, float* __restrict__ out) {
  int n = threadIdx.x;
  if (n >= 8) return;
  float acc = 0.f;
  for (int i = 0; i < 1000; i++) acc += h[n*1000 + i] * W[i];
  out[n] = lrelu(acc + b[0]);
}

// ------------------------------------------------------------------
// Host orchestration
// ------------------------------------------------------------------
static inline long cdivl(long a, long b) { return (a + b - 1) / b; }

extern "C" void kernel_launch(void* const* d_in, const int* in_sizes, int n_in,
                              void* d_out, int out_size, void* d_ws, size_t ws_size,
                              hipStream_t stream) {
  const float* feat  = (const float*)d_in[0];
  const int*   coors = (const int*)d_in[1];
  const float* w1 = (const float*)d_in[3];  const float* b1 = (const float*)d_in[4];
  const float* w2 = (const float*)d_in[5];  const float* b2 = (const float*)d_in[6];
  const float* w3 = (const float*)d_in[7];  const float* b3 = (const float*)d_in[8];
  const float* w4 = (const float*)d_in[9];  const float* b4 = (const float*)d_in[10];
  const float* w5 = (const float*)d_in[11]; const float* b5 = (const float*)d_in[12];
  const float* w6 = (const float*)d_in[13]; const float* b6 = (const float*)d_in[14];
  const float* w7 = (const float*)d_in[15]; const float* b7 = (const float*)d_in[16];
  const float* W8 = (const float*)d_in[17]; const float* b8 = (const float*)d_in[18];
  const float* W9 = (const float*)d_in[19]; const float* b9 = (const float*)d_in[20];
  const float* W10= (const float*)d_in[21]; const float* b10= (const float*)d_in[22];
  const int NPTS = in_sizes[0];

  // ---- workspace arena ----
  char* base = (char*)d_ws;
  size_t off = 0;
  auto arena = [&](size_t bytes) -> char* {
    char* p = base + off;
    off += (bytes + 255) & ~(size_t)255;
    return p;
  };
  const size_t MASKB = (size_t)8*50*60*65*4 + 1024;               // 6.24 MB
  const size_t ACTB  = (size_t)8*47*57*62*32*2 + 4096;            // 85 MB (+pad for tile overreads)
  float* dense = (float*)arena(MASKB);
  float* mA  = (float*)arena(MASKB);
  float* mB  = (float*)arena(MASKB);
  float* mt1 = (float*)arena(MASKB);
  float* mt2 = (float*)arena(MASKB);
  h16* actA = (h16*)arena(ACTB);
  h16* actB = (h16*)arena(ACTB);
  h16* actC = (h16*)arena(ACTB);
  h16* wr2 = (h16*)arena(64*32*32*2);
  h16* wr3 = (h16*)arena(64*32*32*2);
  h16* wr4 = (h16*)arena(64*32*32*2);
  h16* wr5 = (h16*)arena(64*32*32*2);
  h16* wr6 = (h16*)arena(64*32*32*2);
  h16* wr7 = (h16*)arena(2048*2);
  float* c7a = (float*)arena((size_t)8*21*31*36*4 + 1024);
  float* c7b = (float*)arena((size_t)8*17*27*32*4 + 1024);
  float* h1  = (float*)arena(8*1000*4);
  float* h2  = (float*)arena(8*1000*4);
  (void)ws_size; (void)n_in; (void)out_size;

  // separable 3-pass max-pool helpers (x, then y, then z)
  auto pool3_f32 = [&](const float* in, float* out, int D, int H, int W, int k,
                       const float* pm, const float* gm) {
    long t1 = (long)8*D*H*(W-k+1);
    k_pool1d_f32<<<cdivl(t1,256),256,0,stream>>>(in, mt1, pm, nullptr, 8, D, H, W, 2, k);
    long t2 = (long)8*D*(H-k+1)*(W-k+1);
    k_pool1d_f32<<<cdivl(t2,256),256,0,stream>>>(mt1, mt2, nullptr, nullptr, 8, D, H, W-k+1, 1, k);
    long t3 = (long)8*(D-k+1)*(H-k+1)*(W-k+1);
    k_pool1d_f32<<<cdivl(t3,256),256,0,stream>>>(mt2, out, nullptr, gm, 8, D, H-k+1, W-k+1, 0, k);
  };
  auto pool3_f16 = [&](const h16* in, h16* t1b, h16* t2b, h16* out, int D, int H, int W, int k,
                       const float* pm, const float* gm) {
    long t1 = (long)8*D*H*(W-k+1)*4;
    k_pool1d_f16c<<<cdivl(t1,256),256,0,stream>>>(in, t1b, pm, nullptr, 8, D, H, W, 2, k);
    long t2 = (long)8*D*(H-k+1)*(W-k+1)*4;
    k_pool1d_f16c<<<cdivl(t2,256),256,0,stream>>>(t1b, t2b, nullptr, nullptr, 8, D, H, W-k+1, 1, k);
    long t3 = (long)8*(D-k+1)*(H-k+1)*(W-k+1)*4;
    k_pool1d_f16c<<<cdivl(t3,256),256,0,stream>>>(t2b, out, nullptr, gm, 8, D, H-k+1, W-k+1, 0, k);
  };
  auto conv = [&](const h16* in, h16* out, const h16* wr, const float* bias,
                  const float* m2, int D, int H, int W) {
    long waves = (long)8*(D-3)*(((H-3)+1)/2)*(((W-3)+15)/16);
    k_conv_wmma<<<cdivl(waves*32,256),256,131072,stream>>>(in, out, wr, bias, m2, D, H, W);
  };

  // 1) scatter into dense volume + mask
  hipMemsetAsync(dense, 0, (size_t)8*50*60*65*4, stream);
  hipMemsetAsync(mA,    0, (size_t)8*50*60*65*4, stream);
  k_scatter<<<cdivl(NPTS,256),256,0,stream>>>(feat, coors, dense, mA, NPTS);

  // 2) reorder conv weights to [tap][co][ci] f16
  k_reorder32<<<cdivl(65536,256),256,0,stream>>>(w2, wr2);
  k_reorder32<<<cdivl(65536,256),256,0,stream>>>(w3, wr3);
  k_reorder32<<<cdivl(65536,256),256,0,stream>>>(w4, wr4);
  k_reorder32<<<cdivl(65536,256),256,0,stream>>>(w5, wr5);
  k_reorder32<<<cdivl(65536,256),256,0,stream>>>(w6, wr6);
  k_reorder7 <<<cdivl(2048,256),256,0,stream>>>(w7, wr7);

  // 3) conv1 (1->32): mask pool 4^3 then scalar conv.  dims: (50,60,65) -> (47,57,62)
  pool3_f32(mA, mB, 50, 60, 65, 4, nullptr, nullptr);             // mB @ s1
  { long t = (long)8*47*57*62;
    k_conv1<<<cdivl(t,256),256,0,stream>>>(dense, mB, w1, b1, actA); }

  // 4) conv2 (WMMA): (47,57,62) -> (44,54,59)
  pool3_f32(mB, mA, 47, 57, 62, 4, nullptr, nullptr);             // mA @ s2
  conv(actA, actB, wr2, b2, mA, 47, 57, 62);

  // 5) sparse maxpool k=5: (44,54,59) -> (40,50,55)
  pool3_f32(mA, mB, 44, 54, 59, 5, nullptr, nullptr);             // mB @ s3
  pool3_f16(actB, actC, actA, actB, 44, 54, 59, 5, mA, mB);       // x in actB @ s3

  // 6) conv3: (40,50,55) -> (37,47,52)
  pool3_f32(mB, mA, 40, 50, 55, 4, nullptr, nullptr);             // mA @ s4
  conv(actB, actA, wr3, b3, mA, 40, 50, 55);

  // 7) conv4: (37,47,52) -> (34,44,49)
  pool3_f32(mA, mB, 37, 47, 52, 4, nullptr, nullptr);             // mB @ s5
  conv(actA, actB, wr4, b4, mB, 37, 47, 52);

  // 8) sparse maxpool k=5: (34,44,49) -> (30,40,45)
  pool3_f32(mB, mA, 34, 44, 49, 5, nullptr, nullptr);             // mA @ s6
  pool3_f16(actB, actC, actA, actB, 34, 44, 49, 5, mB, mA);       // x in actB @ s6

  // 9) conv5: (30,40,45) -> (27,37,42)
  pool3_f32(mA, mB, 30, 40, 45, 4, nullptr, nullptr);             // mB @ s7
  conv(actB, actA, wr5, b5, mB, 30, 40, 45);

  // 10) conv6: (27,37,42) -> (24,34,39)
  pool3_f32(mB, mA, 27, 37, 42, 4, nullptr, nullptr);             // mA @ s8
  conv(actA, actB, wr6, b6, mA, 27, 37, 42);

  // 11) conv7 (32->1): (24,34,39) -> (21,31,36), f32 single channel
  pool3_f32(mA, mB, 24, 34, 39, 4, nullptr, nullptr);             // mB @ s9
  { long t = (long)8*21*31*36;
    k_conv7<<<cdivl(t,256),256,0,stream>>>(actB, c7a, wr7, b7, mB, 24, 34, 39); }

  // 12) final sparse maxpool k=5: (21,31,36) -> (17,27,32)
  pool3_f32(mB, mA, 21, 31, 36, 5, nullptr, nullptr);             // mA @ s10
  { // x pool, single channel: x-pass (premask), y-pass, z-pass (gate)
    long t1 = (long)8*21*31*32;
    k_pool1d_f32<<<cdivl(t1,256),256,0,stream>>>(c7a, mt1, mB, nullptr, 8, 21, 31, 36, 2, 5);
    long t2 = (long)8*21*27*32;
    k_pool1d_f32<<<cdivl(t2,256),256,0,stream>>>(mt1, mt2, nullptr, nullptr, 8, 21, 31, 32, 1, 5);
    long t3 = (long)8*17*27*32;
    k_pool1d_f32<<<cdivl(t3,256),256,0,stream>>>(mt2, c7b, nullptr, mA, 8, 21, 27, 32, 0, 5);
  }

  // 13) MLP head: 14688 -> 1000 -> 1000 -> 1  (flatten order matches z,y,x)
  k_fc<<<cdivl(1000,128),128,0,stream>>>(c7b, W8, b8, h1, 14688, 1000);
  k_fc<<<cdivl(1000,128),128,0,stream>>>(h1,  W9, b9, h2, 1000, 1000);
  k_fc_last<<<1,8,0,stream>>>(h2, W10, b10, (float*)d_out);
}
